// GNNModule_36249523978501
// MI455X (gfx1250) — matrix operand order repaired and checked by reading, not compiled, not run on hardware
//
#include <hip/hip_runtime.h>

typedef __attribute__((ext_vector_type(16))) __bf16  v16bf;
typedef __attribute__((ext_vector_type(8)))  __bf16  v8bf;
typedef __attribute__((ext_vector_type(8)))  float   v8f;
typedef __attribute__((ext_vector_type(4)))  unsigned int v4u;
typedef __attribute__((ext_vector_type(8)))  int     v8i;
typedef __attribute__((ext_vector_type(4)))  int     v4i;

constexpr int kN = 50000;
constexpr int kE = 600000;
constexpr int kD = 128;
constexpr int kPanel = kD * kD;           // 16384 bf16 elements = 32 KB

// ---------------------------------------------------------------- WMMA helpers

__device__ __forceinline__ v8f wmma_bf16(v16bf a, v16bf b, v8f c) {
  return __builtin_amdgcn_wmma_f32_16x16x32_bf16(false, a, false, b, (short)0, c,
                                                 false, false);
}

// A fragment (16x32 tile) from one fp32 row in global memory.
// Per ISA 16-bit A layout: lane (kh = lane>>4) holds K = k0..k0+7 and
// k0+16..k0+23 where k0 = kt*32 + kh*8.
__device__ __forceinline__ v16bf frag_row(const float* __restrict__ rp, int k0) {
  union { v16bf v; __bf16 e[16]; } u;
  float4 x0 = *(const float4*)(rp + k0);
  float4 x1 = *(const float4*)(rp + k0 + 4);
  float4 x2 = *(const float4*)(rp + k0 + 16);
  float4 x3 = *(const float4*)(rp + k0 + 20);
  u.e[0]=(__bf16)x0.x;  u.e[1]=(__bf16)x0.y;  u.e[2]=(__bf16)x0.z;  u.e[3]=(__bf16)x0.w;
  u.e[4]=(__bf16)x1.x;  u.e[5]=(__bf16)x1.y;  u.e[6]=(__bf16)x1.z;  u.e[7]=(__bf16)x1.w;
  u.e[8]=(__bf16)x2.x;  u.e[9]=(__bf16)x2.y;  u.e[10]=(__bf16)x2.z; u.e[11]=(__bf16)x2.w;
  u.e[12]=(__bf16)x3.x; u.e[13]=(__bf16)x3.y; u.e[14]=(__bf16)x3.z; u.e[15]=(__bf16)x3.w;
  return u.v;
}

// Same, but value = s1*x + aggr (GINE pre-MLP combine).
__device__ __forceinline__ v16bf frag_node(const float* __restrict__ xr,
                                           const float* __restrict__ ar,
                                           float s1, int k0) {
  union { v16bf v; __bf16 e[16]; } u;
  float4 x0 = *(const float4*)(xr + k0);
  float4 x1 = *(const float4*)(xr + k0 + 4);
  float4 x2 = *(const float4*)(xr + k0 + 16);
  float4 x3 = *(const float4*)(xr + k0 + 20);
  float4 a0 = *(const float4*)(ar + k0);
  float4 a1 = *(const float4*)(ar + k0 + 4);
  float4 a2 = *(const float4*)(ar + k0 + 16);
  float4 a3 = *(const float4*)(ar + k0 + 20);
  u.e[0]=(__bf16)(s1*x0.x+a0.x);  u.e[1]=(__bf16)(s1*x0.y+a0.y);
  u.e[2]=(__bf16)(s1*x0.z+a0.z);  u.e[3]=(__bf16)(s1*x0.w+a0.w);
  u.e[4]=(__bf16)(s1*x1.x+a1.x);  u.e[5]=(__bf16)(s1*x1.y+a1.y);
  u.e[6]=(__bf16)(s1*x1.z+a1.z);  u.e[7]=(__bf16)(s1*x1.w+a1.w);
  u.e[8]=(__bf16)(s1*x2.x+a2.x);  u.e[9]=(__bf16)(s1*x2.y+a2.y);
  u.e[10]=(__bf16)(s1*x2.z+a2.z); u.e[11]=(__bf16)(s1*x2.w+a2.w);
  u.e[12]=(__bf16)(s1*x3.x+a3.x); u.e[13]=(__bf16)(s1*x3.y+a3.y);
  u.e[14]=(__bf16)(s1*x3.z+a3.z); u.e[15]=(__bf16)(s1*x3.w+a3.w);
  return u.v;
}

// A fragment from LDS scratch stored [row][k] bf16 (split-8 K pattern).
__device__ __forceinline__ v16bf frag_ldsA(const __bf16* p) {
  union { v16bf v; v8bf h[2]; } u;
  u.h[0] = *(const v8bf*)(p);
  u.h[1] = *(const v8bf*)(p + 16);
  return u.v;
}

// B fragment (32x16) from LDS weights stored transposed [n][k]:
// lane holds column n = nt*16 + (lane&15), K = kt*32 + kh*8 .. +7 and +16 .. +23.
__device__ __forceinline__ v16bf frag_ldsB(const __bf16* p) {
  union { v16bf v; v8bf h[2]; } u;
  u.h[0] = *(const v8bf*)(p);
  u.h[1] = *(const v8bf*)(p + 16);
  return u.v;
}

// ------------------------------------------------------- TDM panel load to LDS
// DMA `nbytes` (multiple of 8, here 32 KB) from global `gptr` into LDS at byte
// offset `lds_off`, as a 1-row tile of 8-byte elements (D# groups per CDNA5
// ISA ch.8). Issue from ONE wave (uniform branch); EXEC is ignored by TDM.
__device__ __forceinline__ void tdm_load_panel(unsigned lds_off,
                                               const void* gptr,
                                               unsigned nbytes) {
  unsigned long long ga = (unsigned long long)(size_t)gptr;
  unsigned n8 = nbytes >> 3;                 // elements of data_size=8B
  v4u g0;
  g0[0] = 1u;                                // count=1, user descriptor
  g0[1] = lds_off;                           // lds_addr (bytes)
  g0[2] = (unsigned)(ga & 0xffffffffu);      // global_addr[31:0]
  g0[3] = (unsigned)((ga >> 32) & 0x01ffffffu)   // global_addr[56:32]
        | (2u << 30);                        // type=2 ("image") bits 127:126
  v8i g1;
  g1[0] = (int)(3u << 16);                   // data_size=3 (8 bytes); mask=0
  g1[1] = (int)((n8 & 0xffffu) << 16);       // tensor_dim0[15:0]  (bits 63:48)
  g1[2] = (int)(((n8 >> 16) & 0xffffu)       // tensor_dim0[31:16] (bits 79:64)
        | (1u << 16));                       // tensor_dim1 = 1    (bits 95:80)
  g1[3] = (int)((n8 & 0xffffu) << 16);       // tile_dim0          (bits 127:112)
  g1[4] = 1;                                 // tile_dim1 = 1      (bits 143:128)
  g1[5] = (int)n8;                           // tensor_dim0_stride[31:0]
  g1[6] = 0;                                 // stride[47:32]=0; dim1_stride=0
  g1[7] = 0;
  v4i z4 = {0, 0, 0, 0};
  v8i z8 = {0, 0, 0, 0, 0, 0, 0, 0};
  // 6-arg toolchain variant: (g0, g1, g2, g3, g4, cpol)
  __builtin_amdgcn_tensor_load_to_lds(g0, g1, z4, z4, z8, 0);
}

__device__ __forceinline__ unsigned lds_offset_of(const void* p) {
  // generic LDS address = aperture(high 32b) | byte offset(low 32b)
  return (unsigned)(size_t)p;
}

// --------------------------------------------- weight prep: bf16, [n][k] panels
// W: panels x [k(128) x n(128)] f32 row-major  ->  out: panels x [n][k] bf16
__global__ __launch_bounds__(256) void k_prep_wT(const float* __restrict__ W,
                                                 __bf16* __restrict__ out) {
  size_t p = blockIdx.y;
  int i = blockIdx.x * 256 + threadIdx.x;    // 0..16383, coalesced over n
  int k = i >> 7, n = i & 127;
  out[p * kPanel + (size_t)n * kD + k] = (__bf16)W[p * kPanel + i];
}

// ---------------------------------------------------------- message + scatter
// msg = relu(x[src] + edge_attr); aggr[dst] += msg  (f32 global atomics)
__global__ __launch_bounds__(256) void k_gather_scatter(
    const float* __restrict__ x, const float* __restrict__ ea,
    const int* __restrict__ src, const int* __restrict__ dst,
    float* __restrict__ aggr) {
  long long gid = (long long)blockIdx.x * blockDim.x + threadIdx.x;
  int e = (int)(gid >> 5);
  if (e >= kE) return;
  int c = ((int)gid & 31) * 4;
  int s = src[e], d = dst[e];
  float4 xv = *(const float4*)(x + (size_t)s * kD + c);
  float4 ev = *(const float4*)(ea + (size_t)e * kD + c);
  float4 m;
  m.x = fmaxf(xv.x + ev.x, 0.f);
  m.y = fmaxf(xv.y + ev.y, 0.f);
  m.z = fmaxf(xv.z + ev.z, 0.f);
  m.w = fmaxf(xv.w + ev.w, 0.f);
  float* o = aggr + (size_t)d * kD + c;
  __hip_atomic_fetch_add(o + 0, m.x, __ATOMIC_RELAXED, __HIP_MEMORY_SCOPE_AGENT);
  __hip_atomic_fetch_add(o + 1, m.y, __ATOMIC_RELAXED, __HIP_MEMORY_SCOPE_AGENT);
  __hip_atomic_fetch_add(o + 2, m.z, __ATOMIC_RELAXED, __HIP_MEMORY_SCOPE_AGENT);
  __hip_atomic_fetch_add(o + 3, m.w, __ATOMIC_RELAXED, __HIP_MEMORY_SCOPE_AGENT);
}

// ------------------------------------------------------------- node MLP (WMMA)
// h = relu(((1+eps)x + aggr) @ W1 + b1) @ W2 + b2    -> hout
// W panels arrive pre-transposed bf16 [n][k]; staged into LDS by the TDM.
__global__ __launch_bounds__(128) void k_node_mlp(
    const float* __restrict__ x, const float* __restrict__ aggr,
    const __bf16* __restrict__ w1t, const float* __restrict__ b1,
    const __bf16* __restrict__ w2t, const float* __restrict__ b2,
    const float* __restrict__ epsp, float* __restrict__ hout, int nTiles) {
  __shared__ __align__(16) __bf16 sW2T[kPanel];  // [n][k], resident
  __shared__ __align__(16) __bf16 sW1T[kPanel];  // [n][k]; aliases scratch
  const int tid  = threadIdx.x;
  const int lane = tid & 31, wave = tid >> 5;
  const int row  = lane & 15, kh = lane >> 4;
  const float s1 = 1.0f + epsp[0];

  if (wave == 0) {                               // TDM: W2^T panel, once
    tdm_load_panel(lds_offset_of(sW2T), w2t, kPanel * 2);
    __builtin_amdgcn_s_wait_tensorcnt(0);
  }
  float b1v[8], b2v[8];
#pragma unroll
  for (int nt = 0; nt < 8; ++nt) { b1v[nt] = b1[nt * 16 + row]; b2v[nt] = b2[nt * 16 + row]; }
  __bf16* scratch = sW1T + wave * (16 * kD);

  for (int t0 = blockIdx.x * 4; t0 < nTiles; t0 += gridDim.x * 4) {
    __syncthreads();                             // scratch quiesced (prev iter)
    if (wave == 0) {                             // TDM: W1^T panel
      tdm_load_panel(lds_offset_of(sW1T), w1t, kPanel * 2);
      __builtin_amdgcn_s_wait_tensorcnt(0);
    }
    __syncthreads();
    const int t = t0 + wave;
    const bool active = t < nTiles;              // wave-uniform
    v8f acc[8] = {};
    if (active) {
      const float* xr = x    + (size_t)(t * 16 + row) * kD;
      const float* ar = aggr + (size_t)(t * 16 + row) * kD;
      v16bf af[4];
#pragma unroll
      for (int kt = 0; kt < 4; ++kt) af[kt] = frag_node(xr, ar, s1, kt * 32 + kh * 8);
#pragma unroll
      for (int nt = 0; nt < 8; ++nt)
#pragma unroll
        for (int kt = 0; kt < 4; ++kt)
          acc[nt] = wmma_bf16(af[kt],
              frag_ldsB(sW1T + (nt * 16 + row) * kD + kt * 32 + kh * 8), acc[nt]);
    }
    __syncthreads();
    if (active) {                                // relu(. + b1) -> scratch [row][k]
#pragma unroll
      for (int nt = 0; nt < 8; ++nt)
#pragma unroll
        for (int v = 0; v < 8; ++v)
          scratch[(v + kh * 8) * kD + nt * 16 + row] =
              (__bf16)fmaxf(acc[nt][v] + b1v[nt], 0.f);
    }
    __syncthreads();
    if (active) {
      v16bf a2[4];
#pragma unroll
      for (int kt = 0; kt < 4; ++kt)
        a2[kt] = frag_ldsA(scratch + row * kD + kt * 32 + kh * 8);
      v8f acc2[8] = {};
#pragma unroll
      for (int nt = 0; nt < 8; ++nt)
#pragma unroll
        for (int kt = 0; kt < 4; ++kt)
          acc2[nt] = wmma_bf16(a2[kt],
              frag_ldsB(sW2T + (nt * 16 + row) * kD + kt * 32 + kh * 8), acc2[nt]);
      float* hr = hout + (size_t)t * 16 * kD;
#pragma unroll
      for (int nt = 0; nt < 8; ++nt)
#pragma unroll
        for (int v = 0; v < 8; ++v)
          hr[(size_t)(v + kh * 8) * kD + nt * 16 + row] = acc2[nt][v] + b2v[nt];
    }
  }
}

// ------------------------------------------------------------------- BN stats
__global__ __launch_bounds__(128) void k_bn_stats(const float* __restrict__ h,
                                                  float* __restrict__ stats) {
  int f = threadIdx.x;
  float s = 0.f, s2 = 0.f;
  for (int r = blockIdx.x; r < kN; r += gridDim.x) {
    float v = h[(size_t)r * kD + f];
    s += v; s2 += v * v;
  }
  __hip_atomic_fetch_add(&stats[f],      s,  __ATOMIC_RELAXED, __HIP_MEMORY_SCOPE_AGENT);
  __hip_atomic_fetch_add(&stats[kD + f], s2, __ATOMIC_RELAXED, __HIP_MEMORY_SCOPE_AGENT);
}

// ---------------------------------------------------- BN apply + residual half
__global__ __launch_bounds__(256) void k_bn_apply(
    const float* __restrict__ xin, const float* __restrict__ h,
    const float* __restrict__ stats, const float* __restrict__ gamma,
    const float* __restrict__ beta, float* __restrict__ xout) {
  long long i = (long long)blockIdx.x * blockDim.x + threadIdx.x;
  if (i >= (long long)kN * kD) return;
  int f = (int)(i & (kD - 1));
  float mu  = stats[f] * (1.f / kN);
  float var = stats[kD + f] * (1.f / kN) - mu * mu;
  float hv  = (h[i] - mu) * rsqrtf(var + 1e-5f) * gamma[f] + beta[f];
  xout[i] = (xin[i] + fmaxf(hv, 0.f)) * 0.5f;
}

// ------------------------------------------------------------- edge MLP (WMMA)
// e = relu([x[s],x[d],ea] @ eW1 + eb1) @ eW2 + eb2 ; eaout = ea + 0.5*e
// ew1t = three [n][k] bf16 panels (K chunks); ew2t = one panel.
__global__ __launch_bounds__(128) void k_edge_mlp(
    const float* __restrict__ x, const float* __restrict__ ea,
    const int* __restrict__ src, const int* __restrict__ dst,
    const __bf16* __restrict__ ew1t, const float* __restrict__ eb1,
    const __bf16* __restrict__ ew2t, const float* __restrict__ eb2,
    float* __restrict__ eaout, int nTiles) {
  __shared__ __align__(16) __bf16 sW2T[kPanel];  // [n][k], resident
  __shared__ __align__(16) __bf16 sW1T[kPanel];  // eW1 K-chunk; aliases scratch
  const int tid  = threadIdx.x;
  const int lane = tid & 31, wave = tid >> 5;
  const int row  = lane & 15, kh = lane >> 4;

  if (wave == 0) {                               // TDM: eW2^T panel, once
    tdm_load_panel(lds_offset_of(sW2T), ew2t, kPanel * 2);
    __builtin_amdgcn_s_wait_tensorcnt(0);
  }
  float b1v[8], b2v[8];
#pragma unroll
  for (int nt = 0; nt < 8; ++nt) { b1v[nt] = eb1[nt * 16 + row]; b2v[nt] = eb2[nt * 16 + row]; }
  __bf16* scratch = sW1T + wave * (16 * kD);

  for (int t0 = blockIdx.x * 4; t0 < nTiles; t0 += gridDim.x * 4) {
    const int t = t0 + wave;
    const bool active = t < nTiles;              // wave-uniform
    int e = t * 16 + row, s = 0, d = 0;
    if (active) { s = src[e]; d = dst[e]; }
    v8f acc[8] = {};
    for (int c = 0; c < 3; ++c) {                // K = 384 in three 128-chunks
      __syncthreads();                           // chunk/scratch quiesced
      if (wave == 0) {                           // TDM: eW1^T chunk c
        tdm_load_panel(lds_offset_of(sW1T), ew1t + (size_t)c * kPanel, kPanel * 2);
        __builtin_amdgcn_s_wait_tensorcnt(0);
      }
      __syncthreads();
      if (active) {
        const float* rp = (c == 0) ? x + (size_t)s * kD
                        : (c == 1) ? x + (size_t)d * kD
                                   : ea + (size_t)e * kD;
        v16bf af[4];
#pragma unroll
        for (int kt = 0; kt < 4; ++kt) af[kt] = frag_row(rp, kt * 32 + kh * 8);
#pragma unroll
        for (int nt = 0; nt < 8; ++nt)
#pragma unroll
          for (int kt = 0; kt < 4; ++kt)
            acc[nt] = wmma_bf16(af[kt],
                frag_ldsB(sW1T + (nt * 16 + row) * kD + kt * 32 + kh * 8), acc[nt]);
      }
    }
    __syncthreads();
    if (active) {
#pragma unroll
      for (int nt = 0; nt < 8; ++nt)
#pragma unroll
        for (int v = 0; v < 8; ++v)
          scratch[(v + kh * 8) * kD + nt * 16 + row] =
              (__bf16)fmaxf(acc[nt][v] + b1v[nt], 0.f);
    }
    __syncthreads();
    if (active) {
      v16bf a2[4];
#pragma unroll
      for (int kt = 0; kt < 4; ++kt)
        a2[kt] = frag_ldsA(scratch + row * kD + kt * 32 + kh * 8);
      v8f acc2[8] = {};
#pragma unroll
      for (int nt = 0; nt < 8; ++nt)
#pragma unroll
        for (int kt = 0; kt < 4; ++kt)
          acc2[nt] = wmma_bf16(a2[kt],
              frag_ldsB(sW2T + (nt * 16 + row) * kD + kt * 32 + kh * 8), acc2[nt]);
#pragma unroll
      for (int nt = 0; nt < 8; ++nt)
#pragma unroll
        for (int v = 0; v < 8; ++v) {
          size_t idx = (size_t)(t * 16 + v + kh * 8) * kD + nt * 16 + row;
          eaout[idx] = ea[idx] + 0.5f * (acc2[nt][v] + b2v[nt]);
        }
    }
  }
}

// -------------------------------------------------------------------- launcher
extern "C" void kernel_launch(void* const* d_in, const int* in_sizes, int n_in,
                              void* d_out, int out_size, void* d_ws, size_t ws_size,
                              hipStream_t stream) {
  const float* x     = (const float*)d_in[0];
  const int*   ei    = (const int*)d_in[1];   // [2,E]: src then dst
  const float* ea    = (const float*)d_in[2];
  const float* W1    = (const float*)d_in[3];
  const float* b1    = (const float*)d_in[4];
  const float* W2    = (const float*)d_in[5];
  const float* b2    = (const float*)d_in[6];
  const float* eps   = (const float*)d_in[7];
  const float* eW1   = (const float*)d_in[8];
  const float* eb1   = (const float*)d_in[9];
  const float* eW2   = (const float*)d_in[10];
  const float* eb2   = (const float*)d_in[11];
  const float* gamma = (const float*)d_in[12];
  const float* beta  = (const float*)d_in[13];
  const int* srcp = ei;
  const int* dstp = ei + kE;

  float* xout_final = (float*)d_out;                   // [N,D]
  float* eaout      = (float*)d_out + (size_t)kN * kD; // [E,D]

  float*  aggr  = (float*)d_ws;                        // N*D
  float*  hbuf  = aggr + (size_t)kN * kD;              // N*D
  float*  xA    = hbuf + (size_t)kN * kD;              // N*D (x after layer 0)
  float*  stats = xA + (size_t)kN * kD;                // 2*D
  __bf16* wbf   = (__bf16*)(stats + 2 * kD);           // 8 panels x 16384 bf16
  // panel indices: 0,1 = W1T l0,l1 ; 2,3 = W2T l0,l1 ; 4..6 = eW1T chunks l1 ;
  // 7 = eW2T l1
  __bf16* pW1  = wbf;
  __bf16* pW2  = wbf + 2 * (size_t)kPanel;
  __bf16* pEW1 = wbf + 4 * (size_t)kPanel;
  __bf16* pEW2 = wbf + 7 * (size_t)kPanel;

  constexpr int nodeTiles = kN / 16;                   // 3125
  constexpr int edgeTiles = kE / 16;                   // 37500
  const int nodeGrid   = (nodeTiles + 3) / 4;          // 4 tiles (waves) / block
  const int edgeGrid   = (edgeTiles + 3) / 4;
  const int gatherGrid = (kE * 32 + 255) / 256;
  const int applyGrid  = (int)(((long long)kN * kD + 255) / 256);

  // Pre-transpose + bf16-convert all weight panels (layer-0 edge MLP is dead
  // in the reference: its edge_attr_new is overwritten, so only layer-1 eW).
  k_prep_wT<<<dim3(64, 2), 256, 0, stream>>>(W1, pW1);
  k_prep_wT<<<dim3(64, 2), 256, 0, stream>>>(W2, pW2);
  k_prep_wT<<<dim3(64, 3), 256, 0, stream>>>(eW1 + (size_t)1 * 3 * kD * kD, pEW1);
  k_prep_wT<<<dim3(64, 1), 256, 0, stream>>>(eW2 + (size_t)1 * kD * kD, pEW2);

  for (int l = 0; l < 2; ++l) {
    const float* xin = (l == 0) ? x : xA;
    float*       xo  = (l == 0) ? xA : xout_final;

    (void)hipMemsetAsync(aggr, 0, (size_t)kN * kD * sizeof(float), stream);
    (void)hipMemsetAsync(stats, 0, 2 * kD * sizeof(float), stream);

    k_gather_scatter<<<gatherGrid, 256, 0, stream>>>(xin, ea, srcp, dstp, aggr);
    k_node_mlp<<<nodeGrid, 128, 0, stream>>>(xin, aggr,
        pW1 + (size_t)l * kPanel, b1 + l * kD,
        pW2 + (size_t)l * kPanel, b2 + l * kD,
        eps + l, hbuf, nodeTiles);
    k_bn_stats<<<512, 128, 0, stream>>>(hbuf, stats);
    k_bn_apply<<<applyGrid, 256, 0, stream>>>(xin, hbuf, stats,
        gamma + l * kD, beta + l * kD, xo);

    if (l == 1)
      k_edge_mlp<<<edgeGrid, 128, 0, stream>>>(xo, ea, srcp, dstp,
          pEW1, eb1 + l * kD, pEW2, eb2 + l * kD, eaout, edgeTiles);
  }
}
